// BAVectorQuantize_1486058685071
// MI455X (gfx1250) — matrix operand order repaired and checked by reading, not compiled, not run on hardware
//
#include <hip/hip_runtime.h>

// ---------------------------------------------------------------------------
// BA vector-quantize forward for MI455X (gfx1250, wave32, WMMA bf16 + async LDS)
// x:[8,2048,512] f32, codebook:[4096,512] f32, pi:[4096] f32
// out: quantized [16384*512] f32 | indices [16384] f32 | loss [1] f32
// ---------------------------------------------------------------------------

#define N_TOK   16384
#define D_DIM   512
#define K_CB    4096
#define BETA    0.5f
#define EPS_Q   1e-20f

typedef __attribute__((ext_vector_type(16))) __bf16 v16bf;
typedef __attribute__((ext_vector_type(8)))  float  v8f;

union BFrag { v16bf v; unsigned int w[8]; unsigned short u[16]; };

__device__ __forceinline__ unsigned short f2bf(float f) {
    unsigned int x = __float_as_uint(f);
    x += 0x7FFFu + ((x >> 16) & 1u);      // round-to-nearest-even
    return (unsigned short)(x >> 16);
}

// ---------------- fp32 -> bf16 bulk convert (8 elems / thread) -------------
__global__ __launch_bounds__(256) void cvt_bf16_kernel(const float* __restrict__ in,
                                                       unsigned short* __restrict__ out) {
    const int i = blockIdx.x * 256 + threadIdx.x;       // one uint4 (8 bf16) per thread
    const float4* p = (const float4*)in + (size_t)i * 2;
    const float4 a = p[0], b = p[1];
    union { unsigned short u[8]; uint4 v; } r;
    r.u[0] = f2bf(a.x); r.u[1] = f2bf(a.y); r.u[2] = f2bf(a.z); r.u[3] = f2bf(a.w);
    r.u[4] = f2bf(b.x); r.u[5] = f2bf(b.y); r.u[6] = f2bf(b.z); r.u[7] = f2bf(b.w);
    ((uint4*)out)[i] = r.v;
}

// ---------------- row squared-norm (one wave per row) ----------------------
__global__ __launch_bounds__(256) void rowsq_kernel(const float* __restrict__ x,
                                                    float* __restrict__ out) {
    const int wave = threadIdx.x >> 5, lane = threadIdx.x & 31;
    const int row  = blockIdx.x * 8 + wave;
    const float* p = x + (size_t)row * D_DIM;
    float s = 0.f;
    #pragma unroll 4
    for (int i = lane; i < D_DIM; i += 32) { float v = p[i]; s += v * v; }
    #pragma unroll
    for (int o = 16; o; o >>= 1) s += __shfl_xor(s, o, 32);
    if (!lane) out[row] = s;
}

// ---------------- logQ0 = log(pi / sum(pi)) --------------------------------
__global__ __launch_bounds__(256) void logpi_kernel(const float* __restrict__ pi,
                                                    float* __restrict__ lq) {
    __shared__ float red[256];
    __shared__ float tot;
    float s = 0.f;
    for (int k = threadIdx.x; k < K_CB; k += 256) s += pi[k];
    red[threadIdx.x] = s; __syncthreads();
    for (int o = 128; o; o >>= 1) {
        if (threadIdx.x < o) red[threadIdx.x] += red[threadIdx.x + o];
        __syncthreads();
    }
    if (threadIdx.x == 0) tot = red[0];
    __syncthreads();
    const float inv = 1.0f / tot;
    for (int k = threadIdx.x; k < K_CB; k += 256)
        lq[k] = logf(fmaxf(pi[k] * inv, EPS_Q));
}

// ---------------- distance GEMM: d = zsq + csq - 2 z.c^T -------------------
// bf16 inputs pre-converted. Block tile: 128 tokens x 64 codes, 8 waves,
// each wave 32x32 via 2x2 WMMA. Tiles staged with GLOBAL_LOAD_ASYNC_TO_LDS.
#define SAU 40   // LDS row stride in ushorts (80 B: 16B-aligned, odd 16B-chunk stride)
__global__ __launch_bounds__(256) void gemm_dist_kernel(
    const unsigned short* __restrict__ zb, const unsigned short* __restrict__ cbb,
    const float* __restrict__ zsq, const float* __restrict__ csq,
    float* __restrict__ dmat)
{
    __shared__ unsigned short lA[128 * SAU];  // [token][k] bf16
    __shared__ unsigned short lB[64  * SAU];  // [code ][k] bf16
    const int t    = threadIdx.x;
    const int wave = t >> 5, lane = t & 31;
    const int wr = wave & 3;          // token-row group (4 x 32)
    const int wc = wave >> 2;         // code-col group (2 x 32)
    const int rowBase = blockIdx.y * 128;
    const int colBase = blockIdx.x * 64;
    const int half = lane >> 4;       // 0: lanes 0-15, 1: lanes 16-31
    const int lr   = lane & 15;

    // LDS byte offsets (generic->LDS: low 32 bits are the workgroup LDS offset)
    const unsigned ldsA = (unsigned)(size_t)&lA[0];
    const unsigned ldsB = (unsigned)(size_t)&lB[0];

    // per-thread staging coordinates (16-byte chunks = 8 bf16)
    const int ar0 = t >> 2,          ac0 = (t & 3) << 3;          // A chunk 0
    const int ar1 = (t + 256) >> 2,  ac1 = ac0;                   // A chunk 1
    const int br  = t >> 2,          bc  = ac0;                   // B chunk

    v8f zero = {};
    v8f acc[2][2];
    acc[0][0] = zero; acc[0][1] = zero; acc[1][0] = zero; acc[1][1] = zero;

    for (int kk = 0; kk < D_DIM; kk += 32) {
        __syncthreads();   // previous stage fully consumed before LDS overwrite
        {
            const unsigned short* g0 = zb  + (size_t)(rowBase + ar0) * D_DIM + kk + ac0;
            const unsigned short* g1 = zb  + (size_t)(rowBase + ar1) * D_DIM + kk + ac1;
            const unsigned short* g2 = cbb + (size_t)(colBase + br ) * D_DIM + kk + bc;
            const unsigned l0 = ldsA + (unsigned)(ar0 * SAU * 2 + ((t & 3) << 4));
            const unsigned l1 = ldsA + (unsigned)(ar1 * SAU * 2 + ((t & 3) << 4));
            const unsigned l2 = ldsB + (unsigned)(br  * SAU * 2 + ((t & 3) << 4));
            asm volatile("global_load_async_to_lds_b128 %0, %1, off"
                         :: "v"(l0), "v"(g0) : "memory");
            asm volatile("global_load_async_to_lds_b128 %0, %1, off"
                         :: "v"(l1), "v"(g1) : "memory");
            asm volatile("global_load_async_to_lds_b128 %0, %1, off"
                         :: "v"(l2), "v"(g2) : "memory");
        }
        asm volatile("s_wait_asynccnt 0x0" ::: "memory");
        __syncthreads();   // all waves' async copies visible

        // A fragments: 16x32 bf16 per ISA layout (lane<16: K 0-7/16-23; lane>=16: +8)
        BFrag a[2], b[2];
        #pragma unroll
        for (int i = 0; i < 2; ++i) {
            const int row = wr * 32 + i * 16 + lr;
            const int kb  = half * 8;
            #pragma unroll
            for (int g = 0; g < 8; ++g) {
                const int k = kb + ((g < 4) ? (2 * g) : (8 + 2 * g));
                a[i].w[g] = *(const unsigned int*)&lA[row * SAU + k];
            }
        }
        // B fragments: 32x16 bf16 (lane = column; same K layout as A: lane<16: K 0-7/16-23; lane>=16: +8)
        #pragma unroll
        for (int j = 0; j < 2; ++j) {
            const int col = wc * 32 + j * 16 + lr;
            const int kb  = half * 8;
            #pragma unroll
            for (int g = 0; g < 8; ++g) {
                const int k = kb + ((g < 4) ? (2 * g) : (8 + 2 * g));
                b[j].w[g] = *(const unsigned int*)&lB[col * SAU + k];
            }
        }
        #pragma unroll
        for (int i = 0; i < 2; ++i)
            #pragma unroll
            for (int j = 0; j < 2; ++j)
                acc[i][j] = __builtin_amdgcn_wmma_f32_16x16x32_bf16(
                    false, a[i].v, false, b[j].v, (short)0, acc[i][j], false, false);
    }

    // epilogue: d = zsq + csq - 2*s, clip, store
    #pragma unroll
    for (int i = 0; i < 2; ++i) {
        #pragma unroll
        for (int j = 0; j < 2; ++j) {
            const int col   = colBase + wc * 32 + j * 16 + lr;
            const int rbase = rowBase + wr * 32 + i * 16 + half * 8;
            const float cs  = csq[col];
            #pragma unroll
            for (int r = 0; r < 8; ++r) {
                const int row = rbase + r;
                float dd = zsq[row] + cs - 2.0f * acc[i][j][r];
                dd = fmaxf(dd, 1e-8f);
                dmat[(size_t)row * K_CB + col] = dd;
            }
        }
    }
}

// ---------------- BA softmax pass: per-row online softmax + partial Q ------
__global__ __launch_bounds__(256) void ba_pass_kernel(
    const float* __restrict__ dmat, const float* __restrict__ lq,
    float* __restrict__ partials, float* __restrict__ idxOutF,
    int* __restrict__ idxOutI, int writeIdx)
{
    __shared__ float slq[K_CB];
    __shared__ float sm[32];
    __shared__ float sinv[32];
    const int t = threadIdx.x;
    for (int k = t; k < K_CB; k += 256) slq[k] = lq[k];
    __syncthreads();

    const int wave = t >> 5, lane = t & 31;
    const int rowBase = blockIdx.x * 32;

    for (int rr = 0; rr < 4; ++rr) {
        const int r   = wave * 4 + rr;
        const int row = rowBase + r;
        const float* dp = dmat + (size_t)row * K_CB;
        float m = -3.4e38f, l = 0.0f; int mi = 0;
        for (int k = lane; k < K_CB; k += 32) {
            const float v = slq[k] - BETA * dp[k];
            if (v > m) { l = l * __expf(m - v) + 1.0f; m = v; mi = k; }
            else        l += __expf(v - m);
        }
        #pragma unroll
        for (int o = 16; o; o >>= 1) {
            const float om = __shfl_xor(m, o, 32);
            const float ol = __shfl_xor(l, o, 32);
            const int   oi = __shfl_xor(mi, o, 32);
            if (om > m)      { l = l * __expf(m - om) + ol; m = om; mi = oi; }
            else if (om == m){ l += ol; mi = (oi < mi) ? oi : mi; }
            else             { l += ol * __expf(om - m); }
        }
        if (!lane) {
            sm[r] = m; sinv[r] = 1.0f / l;
            if (writeIdx) { idxOutF[row] = (float)mi; idxOutI[row] = mi; }
        }
    }
    __syncthreads();

    // partial Q for this block's 32 rows, fixed summation order (deterministic)
    float* pout = partials + (size_t)blockIdx.x * K_CB;
    for (int k = t; k < K_CB; k += 256) {
        float acc = 0.f;
        #pragma unroll 4
        for (int r = 0; r < 32; ++r) {
            const float v = slq[k] - BETA * dmat[(size_t)(rowBase + r) * K_CB + k];
            acc += __expf(v - sm[r]) * sinv[r];
        }
        pout[k] = acc;
    }
}

// ---------------- reduce partial Q (512 partials per k) --------------------
// mode 0: out = log(clip(mean))   mode 1: out = mean
__global__ __launch_bounds__(256) void reduce_q_kernel(
    const float* __restrict__ partials, float* __restrict__ out, int mode)
{
    const int k = blockIdx.x * 256 + threadIdx.x;
    if (k >= K_CB) return;
    float s = 0.f;
    for (int b = 0; b < N_TOK / 32; ++b) s += partials[(size_t)b * K_CB + k];
    const float q = s * (1.0f / (float)N_TOK);
    out[k] = mode ? q : logf(fmaxf(q, EPS_Q));
}

// ---------------- entropy of Q2 --------------------------------------------
__global__ __launch_bounds__(256) void entropy_kernel(const float* __restrict__ q2,
                                                      float* __restrict__ ent) {
    __shared__ float red[256];
    float s = 0.f;
    for (int k = threadIdx.x; k < K_CB; k += 256) {
        const float q = q2[k];
        s -= q * logf(fmaxf(q, EPS_Q));
    }
    red[threadIdx.x] = s; __syncthreads();
    for (int o = 128; o; o >>= 1) {
        if (threadIdx.x < o) red[threadIdx.x] += red[threadIdx.x + o];
        __syncthreads();
    }
    if (!threadIdx.x) *ent = red[0];
}

// ---------------- gather q, write quantized, per-row SSE -------------------
__global__ __launch_bounds__(256) void gather_kernel(
    const float* __restrict__ z, const float* __restrict__ cb,
    const int* __restrict__ idx, float* __restrict__ qOut,
    float* __restrict__ rowSq)
{
    const int wave = threadIdx.x >> 5, lane = threadIdx.x & 31;
    const int row  = blockIdx.x * 8 + wave;
    const int id   = idx[row];
    const float* zp = z + (size_t)row * D_DIM;
    const float* cp = cb + (size_t)id * D_DIM;
    float* op = qOut + (size_t)row * D_DIM;
    float s = 0.f;
    #pragma unroll 4
    for (int i = lane; i < D_DIM; i += 32) {
        const float q = cp[i];
        const float d = zp[i] - q;
        s += d * d;
        op[i] = q;
    }
    #pragma unroll
    for (int o = 16; o; o >>= 1) s += __shfl_xor(s, o, 32);
    if (!lane) rowSq[row] = s;
}

// ---------------- final loss -----------------------------------------------
__global__ __launch_bounds__(256) void loss_kernel(const float* __restrict__ rowSq,
                                                   const float* __restrict__ ent,
                                                   float* __restrict__ lossOut) {
    __shared__ float red[256];
    float s = 0.f;
    for (int i = threadIdx.x; i < N_TOK; i += 256) s += rowSq[i];
    red[threadIdx.x] = s; __syncthreads();
    for (int o = 128; o; o >>= 1) {
        if (threadIdx.x < o) red[threadIdx.x] += red[threadIdx.x + o];
        __syncthreads();
    }
    if (!threadIdx.x) {
        const float mse = red[0] / ((float)N_TOK * (float)D_DIM);
        *lossOut = 1.25f * mse - 0.01f * (*ent);   // commit(0.25)+codebook(1.0) - 0.01*H(Q)
    }
}

// ---------------------------------------------------------------------------
extern "C" void kernel_launch(void* const* d_in, const int* in_sizes, int n_in,
                              void* d_out, int out_size, void* d_ws, size_t ws_size,
                              hipStream_t stream) {
    const float* z  = (const float*)d_in[0];   // [16384,512]
    const float* cb = (const float*)d_in[1];   // [4096,512]
    const float* pi = (const float*)d_in[2];   // [4096]

    float* out   = (float*)d_out;
    float* qOut  = out;                         // [16384*512]
    float* idxF  = out + (size_t)N_TOK * D_DIM; // [16384]
    float* loss  = idxF + N_TOK;                // [1]

    // workspace layout
    char* base      = (char*)d_ws;
    float* dmat     = (float*)base;                                  // 256 MB
    float* partials = (float*)(base + (size_t)N_TOK * K_CB * 4);     // 8 MB (512 x 4096)
    float* zsq      = partials + (size_t)(N_TOK / 32) * K_CB;
    float* csq      = zsq + N_TOK;
    float* lq       = csq + K_CB;
    float* q2       = lq + K_CB;
    int*   idxI     = (int*)(q2 + K_CB);
    float* rowSq    = (float*)(idxI + N_TOK);
    float* ent      = rowSq + N_TOK;
    unsigned short* zb  = (unsigned short*)(ent + 64);               // 16 MB bf16 z
    unsigned short* cbb = zb + (size_t)N_TOK * D_DIM;                // 4 MB bf16 codebook

    // 1) bf16 pre-convert + norms + prior
    cvt_bf16_kernel<<<(N_TOK * (D_DIM / 8)) / 256, 256, 0, stream>>>(z, zb);
    cvt_bf16_kernel<<<(K_CB * (D_DIM / 8)) / 256, 256, 0, stream>>>(cb, cbb);
    rowsq_kernel<<<N_TOK / 8, 256, 0, stream>>>(z, zsq);
    rowsq_kernel<<<K_CB / 8, 256, 0, stream>>>(cb, csq);
    logpi_kernel<<<1, 256, 0, stream>>>(pi, lq);

    // 2) distance matrix via bf16 WMMA with async-LDS staging
    dim3 gg(K_CB / 64, N_TOK / 128);
    gemm_dist_kernel<<<gg, 256, 0, stream>>>(zb, cbb, zsq, csq, dmat);

    // 3) BA iteration 1: softmax(logQ0 - beta*d), Q1 marginal
    ba_pass_kernel<<<N_TOK / 32, 256, 0, stream>>>(dmat, lq, partials, idxF, idxI, 0);
    reduce_q_kernel<<<K_CB / 256, 256, 0, stream>>>(partials, lq, 0);   // lq = log Q1

    // 4) BA iteration 2: softmax(logQ1 - beta*d), Q2 marginal, argmax -> indices
    ba_pass_kernel<<<N_TOK / 32, 256, 0, stream>>>(dmat, lq, partials, idxF, idxI, 1);
    reduce_q_kernel<<<K_CB / 256, 256, 0, stream>>>(partials, q2, 1);   // q2 = Q2
    entropy_kernel<<<1, 256, 0, stream>>>(q2, ent);

    // 5) gather + quantized + loss
    gather_kernel<<<N_TOK / 8, 256, 0, stream>>>(z, cb, idxI, qOut, rowSq);
    loss_kernel<<<1, 256, 0, stream>>>(rowSq, ent, loss);
}
